// _QuantRNNLayer_37598143709982
// MI455X (gfx1250) — 8-bit matrix operand order repaired and checked by reading, compiled, not run on hardware
//
#include <hip/hip_runtime.h>

typedef int v8i __attribute__((ext_vector_type(8)));

#define RNN_T 512
#define RNN_B 64
#define RNN_I 1024
#define RNN_H 1024
#define BH (RNN_B * RNN_H)

__device__ __forceinline__ v8i zero_v8i() {
  v8i z;
#pragma unroll
  for (int i = 0; i < 8; ++i) z[i] = 0;
  return z;
}

__device__ __forceinline__ float slot_scale(const unsigned* slots, int i) {
  float am = __uint_as_float(slots[i]);
  return fmaxf(am, 1e-8f) * (1.0f / 127.0f);
}

__device__ __forceinline__ float gate_epilogue(float g, float gs, float os, int* qo) {
  float qg = fminf(fmaxf(rintf(g / gs), -128.0f), 127.0f) * gs;
  float t  = tanhf(qg);
  float q  = fminf(fmaxf(rintf(t / os), -128.0f), 127.0f);
  *qo = (int)q;
  return q * os;
}

__global__ void absmax_kernel(const float* __restrict__ src, long long n,
                              unsigned* __restrict__ slot) {
  __shared__ float smem[256];
  float m = 0.0f;
  for (long long i = (long long)blockIdx.x * blockDim.x + threadIdx.x; i < n;
       i += (long long)gridDim.x * blockDim.x)
    m = fmaxf(m, fabsf(src[i]));
  smem[threadIdx.x] = m;
  __syncthreads();
  for (int s = 128; s > 0; s >>= 1) {
    if ((int)threadIdx.x < s)
      smem[threadIdx.x] = fmaxf(smem[threadIdx.x], smem[threadIdx.x + s]);
    __syncthreads();
  }
  if (threadIdx.x == 0) atomicMax(slot, __float_as_uint(smem[0]));
}

__global__ void quant_i8_kernel(const float* __restrict__ src, long long n,
                                const unsigned* __restrict__ slots, int slot,
                                signed char* __restrict__ dst) {
  float inv = 1.0f / slot_scale(slots, slot);
  for (long long i = (long long)blockIdx.x * blockDim.x + threadIdx.x; i < n;
       i += (long long)gridDim.x * blockDim.x) {
    float q = fminf(fmaxf(rintf(src[i] * inv), -128.0f), 127.0f);
    dst[i] = (signed char)(int)q;
  }
}

__global__ void quant_bias_kernel(const float* __restrict__ src, int n,
                                  const unsigned* __restrict__ slots, int slot,
                                  float* __restrict__ dst) {
  float s = slot_scale(slots, slot);
  int i = blockIdx.x * blockDim.x + threadIdx.x;
  if (i < n) {
    float q = fminf(fmaxf(rintf(src[i] / s), -128.0f), 127.0f);
    dst[i] = q * s;
  }
}

__global__ void gemm_pre_kernel(const signed char* __restrict__ qx,
                                const signed char* __restrict__ qw,
                                const unsigned* __restrict__ slots,
                                float* __restrict__ out) {
  const int lane = threadIdx.x & 31;
  const int wave = threadIdx.x >> 5;
  const int m0 = blockIdx.x * 64 + wave * 16;
  const int n0 = blockIdx.y * 64;
  const int r = lane & 15;
  const int half = lane >> 4;

  const signed char* arow = qx + (size_t)(m0 + r) * RNN_I;
  const signed char* bbase = qw + (size_t)(n0 + r) * RNN_I;

  v8i acc[4];
#pragma unroll
  for (int j = 0; j < 4; ++j) acc[j] = zero_v8i();

  for (int k = 0; k < RNN_I; k += 64) {
    v8i a;
#pragma unroll
    for (int g = 0; g < 2; ++g) {
      int4 p = *(const int4*)(arow + k + g * 32 + half * 16);
      a[4 * g] = p.x;
      a[4 * g + 1] = p.y;
      a[4 * g + 2] = p.z;
      a[4 * g + 3] = p.w;
    }
    __builtin_prefetch(arow + k + 64, 0, 1);
#pragma unroll
    for (int j = 0; j < 4; ++j) {
      const signed char* brow = bbase + (size_t)j * 16 * RNN_I;
      int4 q0 = *(const int4*)(brow + k + half * 16);
      int4 q1 = *(const int4*)(brow + k + 32 + half * 16);
      v8i b;
      b[0] = q0.x; b[1] = q0.y; b[2] = q0.z; b[3] = q0.w;
      b[4] = q1.x; b[5] = q1.y; b[6] = q1.z; b[7] = q1.w;
      acc[j] = __builtin_amdgcn_wmma_i32_16x16x64_iu8(true, a, true, b, acc[j],
                                                      false, false);
    }
  }

  const float sc = slot_scale(slots, 0) * slot_scale(slots, 1);
#pragma unroll
  for (int j = 0; j < 4; ++j) {
#pragma unroll
    for (int v = 0; v < 8; ++v) {
      int m = m0 + half * 8 + v;
      int n = n0 + j * 16 + r;
      out[(size_t)m * RNN_H + n] = (float)acc[j][v] * sc;
    }
  }
}

__global__ void step0_kernel(const float* __restrict__ h0,
                             const signed char* __restrict__ qw_hh,
                             const float* __restrict__ qb,
                             const unsigned* __restrict__ slots,
                             const float* __restrict__ gs_p,
                             const float* __restrict__ os_p,
                             float* __restrict__ io0,
                             signed char* __restrict__ qh_out) {
  int idx = blockIdx.x * blockDim.x + threadIdx.x;
  int b = idx >> 10, n = idx & 1023;
  const float* hrow = h0 + (size_t)b * RNN_H;
  const signed char* wrow = qw_hh + (size_t)n * RNN_H;
  float acc = 0.0f;
#pragma unroll 4
  for (int j = 0; j < RNN_H; j += 4) {
    float4 hv = *(const float4*)(hrow + j);
    char4 wv = *(const char4*)(wrow + j);
    acc += hv.x * (float)wv.x + hv.y * (float)wv.y + hv.z * (float)wv.z +
           hv.w * (float)wv.w;
  }
  float s_hh = slot_scale(slots, 2);
  float g = io0[idx] + acc * s_hh + qb[n];
  int qo;
  float h = gate_epilogue(g, *gs_p, *os_p, &qo);
  io0[idx] = h;
  qh_out[idx] = (signed char)qo;
}

__global__ void step_kernel(const signed char* __restrict__ qh_in,
                            const signed char* __restrict__ qw_hh,
                            const float* __restrict__ qb,
                            const unsigned* __restrict__ slots,
                            const float* __restrict__ gs_p,
                            const float* __restrict__ os_p,
                            float* __restrict__ io_t,
                            signed char* __restrict__ qh_out,
                            float* __restrict__ hlast, int write_last) {
  const int lane = threadIdx.x & 31;
  const int wave = threadIdx.x >> 5;
  const int m0 = wave * 16;
  const int n0 = blockIdx.x * 64;
  const int r = lane & 15;
  const int half = lane >> 4;

  const signed char* arow = qh_in + (size_t)(m0 + r) * RNN_H;
  const signed char* bbase = qw_hh + (size_t)(n0 + r) * RNN_H;

  v8i acc[4];
#pragma unroll
  for (int j = 0; j < 4; ++j) acc[j] = zero_v8i();

  for (int k = 0; k < RNN_H; k += 64) {
    v8i a;
#pragma unroll
    for (int g = 0; g < 2; ++g) {
      int4 p = *(const int4*)(arow + k + g * 32 + half * 16);
      a[4 * g] = p.x;
      a[4 * g + 1] = p.y;
      a[4 * g + 2] = p.z;
      a[4 * g + 3] = p.w;
    }
#pragma unroll
    for (int j = 0; j < 4; ++j) {
      const signed char* brow = bbase + (size_t)j * 16 * RNN_H;
      int4 q0 = *(const int4*)(brow + k + half * 16);
      int4 q1 = *(const int4*)(brow + k + 32 + half * 16);
      v8i b;
      b[0] = q0.x; b[1] = q0.y; b[2] = q0.z; b[3] = q0.w;
      b[4] = q1.x; b[5] = q1.y; b[6] = q1.z; b[7] = q1.w;
      acc[j] = __builtin_amdgcn_wmma_i32_16x16x64_iu8(true, a, true, b, acc[j],
                                                      false, false);
    }
  }

  const float gs = *gs_p, os = *os_p;
  const float sc = slot_scale(slots, 2) * os;
#pragma unroll
  for (int j = 0; j < 4; ++j) {
#pragma unroll
    for (int v = 0; v < 8; ++v) {
      int m = m0 + half * 8 + v;
      int n = n0 + j * 16 + r;
      size_t idx = (size_t)m * RNN_H + n;
      float g = io_t[idx] + (float)acc[j][v] * sc + qb[n];
      int qo;
      float h = gate_epilogue(g, gs, os, &qo);
      io_t[idx] = h;
      qh_out[idx] = (signed char)qo;
      if (write_last) hlast[idx] = h;
    }
  }
}

extern "C" void kernel_launch(void* const* d_in, const int* in_sizes, int n_in,
                              void* d_out, int out_size, void* d_ws, size_t ws_size,
                              hipStream_t stream) {
  const float* x    = (const float*)d_in[0];
  const float* h0   = (const float*)d_in[1];
  const float* w_ih = (const float*)d_in[2];
  const float* w_hh = (const float*)d_in[3];
  const float* bias = (const float*)d_in[4];
  const float* gs_p = (const float*)d_in[5];
  const float* os_p = (const float*)d_in[6];
  float* out = (float*)d_out;

  char* ws = (char*)d_ws;
  unsigned* slots   = (unsigned*)ws;
  signed char* qwih = (signed char*)(ws + 256);
  signed char* qwhh = qwih + (size_t)RNN_H * RNN_I;
  float* qb         = (float*)(qwhh + (size_t)RNN_H * RNN_H);
  signed char* qha  = (signed char*)(qb + RNN_H);
  signed char* qhb  = qha + BH;
  signed char* qx   = qhb + BH;

  const long long nX = (long long)RNN_T * RNN_B * RNN_I;
  const long long nWih = (long long)RNN_H * RNN_I;
  const long long nWhh = (long long)RNN_H * RNN_H;

  hipMemsetAsync(slots, 0, 16, stream);

  absmax_kernel<<<1024, 256, 0, stream>>>(x, nX, slots + 0);
  absmax_kernel<<<256, 256, 0, stream>>>(w_ih, nWih, slots + 1);
  absmax_kernel<<<256, 256, 0, stream>>>(w_hh, nWhh, slots + 2);
  absmax_kernel<<<4, 256, 0, stream>>>(bias, RNN_H, slots + 3);

  quant_i8_kernel<<<2048, 256, 0, stream>>>(x, nX, slots, 0, qx);
  quant_i8_kernel<<<512, 256, 0, stream>>>(w_ih, nWih, slots, 1, qwih);
  quant_i8_kernel<<<512, 256, 0, stream>>>(w_hh, nWhh, slots, 2, qwhh);
  quant_bias_kernel<<<4, 256, 0, stream>>>(bias, RNN_H, slots, 3, qb);

  gemm_pre_kernel<<<dim3((RNN_T * RNN_B) / 64, RNN_H / 64), 128, 0, stream>>>(
      qx, qwih, slots, out);

  step0_kernel<<<BH / 256, 256, 0, stream>>>(h0, qwhh, qb, slots, gs_p, os_p,
                                             out, qha);
  float* hlast = out + (size_t)RNN_T * BH;
  for (int t = 1; t < RNN_T; ++t) {
    const signed char* hin = (t & 1) ? qha : qhb;
    signed char* hout = (t & 1) ? qhb : qha;
    step_kernel<<<RNN_H / 64, 128, 0, stream>>>(
        hin, qwhh, qb, slots, gs_p, os_p, out + (size_t)t * BH, hout, hlast,
        (t == RNN_T - 1) ? 1 : 0);
  }
}
